// GraphAttentionLayer_55405078119322
// MI455X (gfx1250) — matrix operand order repaired and checked by reading, not compiled, not run on hardware
//
#include <hip/hip_runtime.h>

// Problem constants (match reference)
#define BATCH 8
#define NN    2048
#define DD    256
#define HH    8
#define HDIM  32
#define KB    128   // keys per attention iteration

typedef __attribute__((ext_vector_type(16))) __bf16 v16bf;
typedef __attribute__((ext_vector_type(8)))  __bf16 v8bf;
typedef __attribute__((ext_vector_type(8)))  float  v8f;

static __device__ __forceinline__ v16bf cat16(v8bf a, v8bf b) {
  return __builtin_shufflevector(a, b, 0,1,2,3,4,5,6,7,8,9,10,11,12,13,14,15);
}
static __device__ __forceinline__ v8bf ldv8(const __bf16* p) {
  return *(const v8bf*)p;
}

#define WMMA_BF16(A, Bm, C) \
  __builtin_amdgcn_wmma_f32_16x16x32_bf16(false, (A), false, (Bm), (short)0, (C), false, false)

// ---------------------------------------------------------------------------
// Kernel 1a: fp32 -> bf16 elementwise convert
// ---------------------------------------------------------------------------
__global__ void k_cvt(const float* __restrict__ src, __bf16* __restrict__ dst, int n) {
  int i = blockIdx.x * blockDim.x + threadIdx.x;
  if (i < n) dst[i] = (__bf16)src[i];
}

// ---------------------------------------------------------------------------
// Kernel 1b: fp32 W[in][out] -> bf16 Wt[out][in]  (256x256)
// ---------------------------------------------------------------------------
__global__ void k_cvt_t(const float* __restrict__ w, __bf16* __restrict__ wt) {
  int i = blockIdx.x * 16 + threadIdx.x;  // input dim
  int o = blockIdx.y * 16 + threadIdx.y;  // output dim
  wt[(size_t)o * DD + i] = (__bf16)w[(size_t)i * DD + o];
}

// ---------------------------------------------------------------------------
// Kernel 2: QKV projection.  One wave per 16x16 tile of [B*N, D].
//   Q,K stored [B,H,N,HD] bf16 ; V stored transposed [B,H,HD,N] bf16.
//   Q (which==0) is pre-scaled by 1/sqrt(HD) so attention skips the scale.
// ---------------------------------------------------------------------------
__global__ void __launch_bounds__(32)
k_qkv(const __bf16* __restrict__ Xb,
      const __bf16* __restrict__ Wqt, const __bf16* __restrict__ Wkt,
      const __bf16* __restrict__ Wvt,
      const float* __restrict__ bq, const float* __restrict__ bk,
      const float* __restrict__ bv,
      __bf16* __restrict__ Qb, __bf16* __restrict__ Kb, __bf16* __restrict__ Vtb) {
  const int mt = blockIdx.x, nt = blockIdx.y, which = blockIdx.z;
  const int lane = threadIdx.x, half = lane >> 4, col = lane & 15;
  const int m0 = mt * 16, n0 = nt * 16;

  const __bf16* Wt   = which == 0 ? Wqt : (which == 1 ? Wkt : Wvt);
  const float*  bias = which == 0 ? bq  : (which == 1 ? bk  : bv);

  v8f c = {};
  const __bf16* arow = Xb + (size_t)(m0 + col) * DD;
  const __bf16* brow = Wt + (size_t)(n0 + col) * DD;
#pragma unroll
  for (int k0 = 0; k0 < DD; k0 += 32) {
    v16bf A  = cat16(ldv8(arow + k0 + 16 * half), ldv8(arow + k0 + 16 * half + 8));
    v16bf Bm = cat16(ldv8(brow + k0 + 16 * half), ldv8(brow + k0 + 16 * half + 8));
    c = WMMA_BF16(A, Bm, c);
  }
  const float bb = bias[n0 + col];
  const float scl = (which == 0) ? 0.17677669529663687f : 1.0f;  // 1/sqrt(HD)
  const int ch = n0 + col;
  const int h = ch >> 5, hd = ch & 31;
#pragma unroll
  for (int v = 0; v < 8; ++v) {
    const int tok  = m0 + v + 8 * half;
    const int bidx = tok >> 11;        // / NN
    const int n    = tok & (NN - 1);
    const float val = (c[v] + bb) * scl;
    if (which == 2) {
      Vtb[((size_t)(bidx * HH + h) * HDIM + hd) * NN + n] = (__bf16)val;
    } else {
      __bf16* dst = (which == 0 ? Qb : Kb);
      dst[((size_t)(bidx * HH + h) * NN + n) * HDIM + hd] = (__bf16)val;
    }
  }
}

// ---------------------------------------------------------------------------
// Kernel 3: flash-style masked attention.
//   One wave per (b, h, 16-query block).  128 keys per iteration:
//   8 score WMMAs -> one amortized online-softmax update -> 8 PV WMMAs.
// ---------------------------------------------------------------------------
__global__ void __launch_bounds__(32)
k_attn(const __bf16* __restrict__ Qb, const __bf16* __restrict__ Kb,
       const __bf16* __restrict__ Vtb, const int* __restrict__ adj,
       __bf16* __restrict__ AOb) {
  const int qt = blockIdx.x, h = blockIdx.y, b = blockIdx.z;
  const int lane = threadIdx.x, half = lane >> 4, col = lane & 15;
  const int n0 = qt * 16;
  const float LOG2E = 1.4426950408889634f;

  const __bf16* Qbh = Qb  + ((size_t)(b * HH + h) * NN) * HDIM;
  const __bf16* Kbh = Kb  + ((size_t)(b * HH + h) * NN) * HDIM;
  const __bf16* Vbh = Vtb + ((size_t)(b * HH + h) * HDIM) * NN;

  // P staging: 16 x 128 bf16, padded stride 136 (272B rows, 16B-aligned slices)
  __shared__ __attribute__((aligned(16))) __bf16 Pl[16][KB + 8];

  // Q A-tile (16x32), loaded once: whole head-dim is one WMMA K-step.
  const __bf16* qp = Qbh + (size_t)(n0 + col) * HDIM;
  v16bf AQ = cat16(ldv8(qp + 16 * half), ldv8(qp + 16 * half + 8));

  v8f o0 = {}, o1 = {};
  float mrow[8], lrow[8];
#pragma unroll
  for (int v = 0; v < 8; ++v) { mrow[v] = -1e30f; lrow[v] = 0.f; }

  for (int mb = 0; mb < NN; mb += KB) {
    if (mb + KB < NN) {
      __builtin_prefetch(Kbh + (size_t)(mb + KB + col) * HDIM, 0, 0);
      __builtin_prefetch(Vbh + (size_t)col * NN + mb + KB, 0, 0);
    }
    // --- scores: S = Q @ K^T  (eight 16-key tiles) ---
    v8f s[8];
#pragma unroll
    for (int t = 0; t < 8; ++t) {
      const __bf16* kp = Kbh + (size_t)(mb + t * 16 + col) * HDIM + 16 * half;
      v16bf Bk = cat16(ldv8(kp), ldv8(kp + 8));
      v8f z = {};
      s[t] = WMMA_BF16(AQ, Bk, z);
    }
    // --- mask + online softmax, amortized over 8 cols/lane per row ---
#pragma unroll
    for (int v = 0; v < 8; ++v) {
      const int r = v + 8 * half;
      const int* arow = adj + ((size_t)b * NN + (n0 + r)) * NN + mb + col;
      int   a[8];
      float x[8];
      float tmax = -1e30f;
#pragma unroll
      for (int t = 0; t < 8; ++t) {
        a[t] = arow[t * 16];
        x[t] = a[t] ? s[t][v] : -1e30f;   // Q was pre-scaled by 1/sqrt(HD)
        tmax = fmaxf(tmax, x[t]);
      }
      tmax = fmaxf(tmax, __shfl_xor(tmax, 1, 32));
      tmax = fmaxf(tmax, __shfl_xor(tmax, 2, 32));
      tmax = fmaxf(tmax, __shfl_xor(tmax, 4, 32));
      tmax = fmaxf(tmax, __shfl_xor(tmax, 8, 32));
      const float mn    = fmaxf(mrow[v], tmax);
      const float alpha = exp2f((mrow[v] - mn) * LOG2E);
      float ps = 0.f;
#pragma unroll
      for (int t = 0; t < 8; ++t) {
        const float p = a[t] ? exp2f((x[t] - mn) * LOG2E) : 0.f;
        ps += p;
        Pl[r][t * 16 + col] = (__bf16)p;   // C-layout -> A-layout via LDS
      }
      ps += __shfl_xor(ps, 1, 32);
      ps += __shfl_xor(ps, 2, 32);
      ps += __shfl_xor(ps, 4, 32);
      ps += __shfl_xor(ps, 8, 32);
      lrow[v] = lrow[v] * alpha + ps;
      mrow[v] = mn;
      o0[v] *= alpha;
      o1[v] *= alpha;
    }
    __syncthreads();
    // --- O += P @ V : 4 K-steps of 32 keys x 2 HD halves ---
#pragma unroll
    for (int kk = 0; kk < KB; kk += 32) {
      const __bf16* pr = &Pl[col][kk];
      v16bf AP = cat16(ldv8(pr + 16 * half), ldv8(pr + 16 * half + 8));
      const __bf16* vp0 = Vbh + (size_t)col * NN + mb + kk + 16 * half;
      v16bf Bv0 = cat16(ldv8(vp0), ldv8(vp0 + 8));
      o0 = WMMA_BF16(AP, Bv0, o0);
      const __bf16* vp1 = Vbh + (size_t)(16 + col) * NN + mb + kk + 16 * half;
      v16bf Bv1 = cat16(ldv8(vp1), ldv8(vp1 + 8));
      o1 = WMMA_BF16(AP, Bv1, o1);
    }
    __syncthreads();
  }
  // normalize and store attention output as bf16 [B, N, D] (heads concatenated)
#pragma unroll
  for (int v = 0; v < 8; ++v) {
    const float inv = 1.0f / lrow[v];
    const int r = v + 8 * half;
    __bf16* dst = AOb + ((size_t)(b * NN + n0 + r)) * DD + h * HDIM;
    dst[col]      = (__bf16)(o0[v] * inv);
    dst[16 + col] = (__bf16)(o1[v] * inv);
  }
}

// ---------------------------------------------------------------------------
// Kernel 4: O-projection + bias + residual + LayerNorm, fused.
//   16 waves per block; wave w computes column tile w of a 16-token strip,
//   then wave w LayerNorms row w.
// ---------------------------------------------------------------------------
__global__ void __launch_bounds__(512)
k_outln(const __bf16* __restrict__ AOb, const __bf16* __restrict__ Wot,
        const float* __restrict__ bo, const float* __restrict__ Xin,
        const float* __restrict__ gamma, const float* __restrict__ beta,
        float* __restrict__ out) {
  const int m0 = blockIdx.x * 16;
  const int wave = threadIdx.x >> 5;
  const int lane = threadIdx.x & 31;
  const int half = lane >> 4, col = lane & 15;
  const int n0 = wave * 16;

  __shared__ float xsh[16][DD];

  v8f c = {};
  const __bf16* arow = AOb + (size_t)(m0 + col) * DD;
  const __bf16* brow = Wot + (size_t)(n0 + col) * DD;
#pragma unroll
  for (int k0 = 0; k0 < DD; k0 += 32) {
    v16bf A  = cat16(ldv8(arow + k0 + 16 * half), ldv8(arow + k0 + 16 * half + 8));
    v16bf Bm = cat16(ldv8(brow + k0 + 16 * half), ldv8(brow + k0 + 16 * half + 8));
    c = WMMA_BF16(A, Bm, c);
  }
  const float bb = bo[n0 + col];
#pragma unroll
  for (int v = 0; v < 8; ++v) {
    const int r = v + 8 * half;
    xsh[r][n0 + col] = c[v] + bb + Xin[(size_t)(m0 + r) * DD + n0 + col];
  }
  __syncthreads();

  // wave w normalizes row w (256 values, 8 per lane)
  const int r = wave;
  float vals[8];
  float sum = 0.f;
#pragma unroll
  for (int j = 0; j < 8; ++j) { vals[j] = xsh[r][lane + 32 * j]; sum += vals[j]; }
  sum += __shfl_xor(sum, 1, 32);
  sum += __shfl_xor(sum, 2, 32);
  sum += __shfl_xor(sum, 4, 32);
  sum += __shfl_xor(sum, 8, 32);
  sum += __shfl_xor(sum, 16, 32);
  const float mu = sum * (1.0f / DD);
  float sq = 0.f;
#pragma unroll
  for (int j = 0; j < 8; ++j) { const float d = vals[j] - mu; sq += d * d; }
  sq += __shfl_xor(sq, 1, 32);
  sq += __shfl_xor(sq, 2, 32);
  sq += __shfl_xor(sq, 4, 32);
  sq += __shfl_xor(sq, 8, 32);
  sq += __shfl_xor(sq, 16, 32);
  const float rstd = rsqrtf(sq * (1.0f / DD) + 1e-5f);
  const int tok = m0 + r;
#pragma unroll
  for (int j = 0; j < 8; ++j) {
    const int ch = lane + 32 * j;
    out[(size_t)tok * DD + ch] = (vals[j] - mu) * rstd * gamma[ch] + beta[ch];
  }
}

// ---------------------------------------------------------------------------
extern "C" void kernel_launch(void* const* d_in, const int* in_sizes, int n_in,
                              void* d_out, int out_size, void* d_ws, size_t ws_size,
                              hipStream_t stream) {
  const float* X     = (const float*)d_in[0];
  const int*   adj   = (const int*)d_in[1];
  const float* Wq    = (const float*)d_in[2];
  const float* bq    = (const float*)d_in[3];
  const float* Wk    = (const float*)d_in[4];
  const float* bk    = (const float*)d_in[5];
  const float* Wv    = (const float*)d_in[6];
  const float* bv    = (const float*)d_in[7];
  const float* Wo    = (const float*)d_in[8];
  const float* bo    = (const float*)d_in[9];
  const float* gamma = (const float*)d_in[10];
  const float* beta  = (const float*)d_in[11];
  float* out = (float*)d_out;

  char* ws = (char*)d_ws;
  size_t off = 0;
  auto alloc = [&](size_t bytes) -> char* {
    char* p = ws + off;
    off += (bytes + 255) & ~(size_t)255;
    return p;
  };
  const size_t nx = (size_t)BATCH * NN * DD;   // 4,194,304 elems
  __bf16* Xb  = (__bf16*)alloc(nx * 2);
  __bf16* Wqt = (__bf16*)alloc((size_t)DD * DD * 2);
  __bf16* Wkt = (__bf16*)alloc((size_t)DD * DD * 2);
  __bf16* Wvt = (__bf16*)alloc((size_t)DD * DD * 2);
  __bf16* Wot = (__bf16*)alloc((size_t)DD * DD * 2);
  __bf16* Qb  = (__bf16*)alloc(nx * 2);        // [B,H,N,HD], pre-scaled
  __bf16* Kb  = (__bf16*)alloc(nx * 2);        // [B,H,N,HD]
  __bf16* Vtb = (__bf16*)alloc(nx * 2);        // [B,H,HD,N]
  __bf16* AOb = (__bf16*)alloc(nx * 2);        // [B,N,D]

  // 1) precision conversion + weight transposes
  k_cvt<<<(int)(nx / 256), 256, 0, stream>>>(X, Xb, (int)nx);
  dim3 tb(16, 16);
  k_cvt_t<<<dim3(16, 16), tb, 0, stream>>>(Wq, Wqt);
  k_cvt_t<<<dim3(16, 16), tb, 0, stream>>>(Wk, Wkt);
  k_cvt_t<<<dim3(16, 16), tb, 0, stream>>>(Wv, Wvt);
  k_cvt_t<<<dim3(16, 16), tb, 0, stream>>>(Wo, Wot);

  // 2) QKV projections (one wave per 16x16 tile, z = which of q/k/v)
  k_qkv<<<dim3((BATCH * NN) / 16, DD / 16, 3), 32, 0, stream>>>(
      Xb, Wqt, Wkt, Wvt, bq, bk, bv, Qb, Kb, Vtb);

  // 3) masked flash attention (one wave per (b,h,16-query block))
  k_attn<<<dim3(NN / 16, HH, BATCH), 32, 0, stream>>>(Qb, Kb, Vtb, adj, AOb);

  // 4) output projection + residual + LayerNorm
  k_outln<<<dim3((BATCH * NN) / 16), 512, 0, stream>>>(
      AOb, Wot, bo, X, gamma, beta, out);
}
